// AttentionBlock_43576738185224
// MI455X (gfx1250) — matrix operand order repaired and checked by reading, not compiled, not run on hardware
//
#include <hip/hip_runtime.h>

// ---------------------------------------------------------------------------
// Types
// ---------------------------------------------------------------------------
typedef __bf16 bf16;
typedef __attribute__((ext_vector_type(8)))  __bf16 v8bf;
typedef __attribute__((ext_vector_type(16))) __bf16 v16bf;
typedef __attribute__((ext_vector_type(8)))  float  v8f;

// ---------------------------------------------------------------------------
// Problem constants (fixed by reference setup_inputs)
// ---------------------------------------------------------------------------
#define S_DIM   256
#define LA      128
#define LB      512
#define DQ      256
#define DKV     265
#define DKVP    288        // padded to multiple of 32 for WMMA K
#define DQK     512
#define DV      512
#define MA      (S_DIM * LA)   // 32768 query rows
#define MB      (S_DIM * LB)   // 131072 key/value rows
#define SCALER  6.25f          // 1/(0.01*sqrt(256))

// ---------------------------------------------------------------------------
// Helpers
// ---------------------------------------------------------------------------
__device__ inline v16bf cat8(v8bf lo, v8bf hi) {
  return __builtin_shufflevector(lo, hi, 0,1,2,3,4,5,6,7,8,9,10,11,12,13,14,15);
}

// ---------------------------------------------------------------------------
// f32 -> bf16 convert with optional channel zero-padding
// ---------------------------------------------------------------------------
__global__ __launch_bounds__(256)
void convert_pad_bf16(const float* __restrict__ src, bf16* __restrict__ dst,
                      long long R, int C, int Cp) {
  long long i = (long long)blockIdx.x * 256 + threadIdx.x;
  long long total = R * (long long)Cp;
  if (i >= total) return;
  long long r = i / Cp;
  int c = (int)(i - r * Cp);
  float v = (c < C) ? src[r * (long long)C + c] : 0.0f;
  dst[i] = (bf16)v;
}

__global__ __launch_bounds__(256)
void zero_f32(float* __restrict__ p, int n) {
  int i = blockIdx.x * 256 + threadIdx.x;
  if (i < n) p[i] = 0.0f;
}

// counts[0] = sum(a_lens), counts[1] = sum(b_lens)   (S == 256 == blockDim)
__global__ __launch_bounds__(256)
void lens_counts(const int* __restrict__ a_lens, const int* __restrict__ b_lens,
                 float* __restrict__ counts) {
  __shared__ int sa[256];
  __shared__ int sb[256];
  int t = threadIdx.x;
  sa[t] = a_lens[t];
  sb[t] = b_lens[t];
  __syncthreads();
  for (int s = 128; s > 0; s >>= 1) {
    if (t < s) { sa[t] += sa[t + s]; sb[t] += sb[t + s]; }
    __syncthreads();
  }
  if (t == 0) { counts[0] = (float)sa[0]; counts[1] = (float)sb[0]; }
}

// ---------------------------------------------------------------------------
// Batched tiled GEMM:  C[b] = X[b] (MxK, bf16) * op(W[b]) + bias
//   TRANSW == false : W is [N,K] row-major (NT gemm, C = X * W^T)
//   TRANSW == true  : W is [K,N] row-major (NN gemm, C = X * W)
// BM = BN = 128, BK = 32, 256 threads = 8 waves, each wave owns 2x4 C-tiles.
// Software-pipelined: next K-tile loaded into registers while WMMAs run.
// bias must be a valid pointer (use a zeroed scratch vector for "no bias").
// ---------------------------------------------------------------------------
#define XS_LD 40   // 32 + 8 pad, keeps 16B alignment per row (80B stride)

template <bool TRANSW>
__global__ __launch_bounds__(256)
void gemm_bf16_wmma(const bf16* __restrict__ X, const bf16* __restrict__ W,
                    const float* __restrict__ bias, float* __restrict__ C,
                    int K, int ldX, int ldW, int ldC,
                    long long sX, long long sW, long long sC) {
  __shared__ bf16 Xs[128 * XS_LD];
  __shared__ bf16 Ws[128 * XS_LD];

  const int b = blockIdx.z;
  X += (long long)b * sX;
  W += (long long)b * sW;
  C += (long long)b * sC;

  const int m0 = blockIdx.y * 128;
  const int n0 = blockIdx.x * 128;
  const int tid  = threadIdx.x;
  const int wave = tid >> 5;
  const int lane = tid & 31;
  const int wrow = wave >> 1;     // 0..3 -> 32 rows each
  const int wcol = wave & 1;      // 0..1 -> 64 cols each
  const int l    = lane & 15;
  const int half = lane >> 4;

  // ---- staging source pointers (k0 = 0) ----
  const int xr = tid >> 1;
  const int xc = (tid & 1) << 4;
  const bf16* xrow = X + (long long)(m0 + xr) * ldX + xc;

  const int wr = tid >> 1;            // NT: N-row per 2 threads
  const int wc = (tid & 1) << 4;
  const int kk = tid >> 3;            // NN: 0..31 K row
  const int nc = (tid & 7) << 4;      // NN: 0..112 N chunk
  const bf16* wbase = TRANSW
      ? (W + (long long)kk * ldW + n0 + nc)
      : (W + (long long)(n0 + wr) * ldW + wc);

  // LDS commit addresses
  bf16* xdst = &Xs[xr * XS_LD + xc];
  bf16* wdst = TRANSW ? &Ws[kk] : &Ws[wr * XS_LD + wc];

  // per-wave fragment base pointers
  const bf16* aB[2];
  const bf16* bB[4];
#pragma unroll
  for (int ti = 0; ti < 2; ++ti)
    aB[ti] = &Xs[(wrow * 32 + ti * 16 + l) * XS_LD];
#pragma unroll
  for (int tj = 0; tj < 4; ++tj)
    bB[tj] = &Ws[(wcol * 64 + tj * 16 + l) * XS_LD];

  auto loadX = [&](int k0, v8bf& a, v8bf& c) {
    const bf16* s = xrow + k0;
    a = *(const v8bf*)(s);
    c = *(const v8bf*)(s + 8);
  };
  auto loadW = [&](int k0, v8bf& a, v8bf& c) {
    const bf16* s = TRANSW ? (wbase + (long long)k0 * ldW) : (wbase + k0);
    a = *(const v8bf*)(s);
    c = *(const v8bf*)(s + 8);
  };
  auto commit = [&](v8bf xa, v8bf xb, v8bf wa, v8bf wb) {
    *(v8bf*)(xdst)     = xa;
    *(v8bf*)(xdst + 8) = xb;
    if (!TRANSW) {
      *(v8bf*)(wdst)     = wa;
      *(v8bf*)(wdst + 8) = wb;
    } else {
#pragma unroll
      for (int i = 0; i < 8; ++i) {
        wdst[(nc + i)     * XS_LD] = wa[i];
        wdst[(nc + 8 + i) * XS_LD] = wb[i];
      }
    }
  };

  v8f acc[2][4];
#pragma unroll
  for (int i = 0; i < 2; ++i)
#pragma unroll
    for (int j = 0; j < 4; ++j)
#pragma unroll
      for (int r = 0; r < 8; ++r) acc[i][j][r] = 0.0f;

  const int nsteps = K >> 5;
  v8bf x0, x1, w0, w1;

  // prologue: fill LDS with tile 0
  loadX(0, x0, x1);
  loadW(0, w0, w1);
  commit(x0, x1, w0, w1);
  __syncthreads();

  for (int s = 0; s < nsteps; ++s) {
    const bool more = (s + 1) < nsteps;

    // L2 pre-warm two tiles ahead
    if (s + 2 < nsteps) {
      __builtin_prefetch(xrow + ((s + 2) << 5), 0, 0);
      __builtin_prefetch(TRANSW ? (wbase + ((long long)(s + 2) << 5) * ldW)
                                : (wbase + ((s + 2) << 5)), 0, 0);
    }
    // issue next tile's global loads (latency hidden behind WMMAs)
    if (more) {
      loadX((s + 1) << 5, x0, x1);
      loadW((s + 1) << 5, w0, w1);
    }

    // ---- load fragments (ISA 7.12.2 layouts) ----
    v16bf afrag[2], bfrag[4];
#pragma unroll
    for (int ti = 0; ti < 2; ++ti) {
      v8bf lo = *(const v8bf*)(aB[ti] + half * 8);        // K 0..7 / 8..15
      v8bf hi = *(const v8bf*)(aB[ti] + 16 + half * 8);   // K 16..23 / 24..31
      afrag[ti] = cat8(lo, hi);
    }
#pragma unroll
    for (int tj = 0; tj < 4; ++tj) {
      v8bf lo = *(const v8bf*)(bB[tj] + half * 8);        // K 0..7 / 8..15
      v8bf hi = *(const v8bf*)(bB[tj] + 16 + half * 8);   // K 16..23 / 24..31
      bfrag[tj] = cat8(lo, hi);
    }
    // ---- 8 WMMAs per wave per K-step ----
#pragma unroll
    for (int ti = 0; ti < 2; ++ti)
#pragma unroll
      for (int tj = 0; tj < 4; ++tj)
        acc[ti][tj] = __builtin_amdgcn_wmma_f32_16x16x32_bf16(
            false, afrag[ti], false, bfrag[tj], (short)0, acc[ti][tj],
            false, false);

    __syncthreads();               // all waves done reading this tile
    if (more) {
      commit(x0, x1, w0, w1);      // overwrite with next tile
      __syncthreads();
    }
  }

  // ---- epilogue: C layout = VGPR r -> M = r + 8*half, N = lane&15 ----
#pragma unroll
  for (int ti = 0; ti < 2; ++ti) {
#pragma unroll
    for (int tj = 0; tj < 4; ++tj) {
      const int col = n0 + wcol * 64 + tj * 16 + l;
      const float bv = bias[col];
      const int rbase = m0 + wrow * 32 + ti * 16 + half * 8;
      float* p = C + (long long)rbase * ldC + col;
#pragma unroll
      for (int r = 0; r < 8; ++r) {
        *p = acc[ti][tj][r] + bv;
        p += ldC;
      }
    }
  }
}

// ---------------------------------------------------------------------------
// Masked BatchNorm statistics: per-channel sum & sum-of-squares via atomics.
// One block handles 128 consecutive rows (always inside a single sample).
// sums layout: [0..D) sum, [D..2D) sumsq
// ---------------------------------------------------------------------------
__global__ __launch_bounds__(256)
void bn_stats(const float* __restrict__ pre, const int* __restrict__ lens,
              int L, int D, float* __restrict__ sums) {
  const int r0  = blockIdx.x * 128;
  const int smp = r0 / L;
  const int pos0 = r0 - smp * L;
  const int len = lens[smp];
  const int t = threadIdx.x;
  for (int c = t; c < D; c += 256) {
    float s = 0.0f, ss = 0.0f;
    for (int i = 0; i < 128; ++i) {
      if (pos0 + i < len) {
        float v = pre[(long long)(r0 + i) * D + c];
        s += v;
        ss += v * v;
      }
    }
    atomicAdd(&sums[c], s);
    atomicAdd(&sums[D + c], ss);
  }
}

// ---------------------------------------------------------------------------
// BN apply + ReLU + optional row L2-normalize -> bf16 (one block per row)
// ---------------------------------------------------------------------------
__global__ __launch_bounds__(256)
void bn_apply_bf16(const float* __restrict__ pre, const float* __restrict__ sums,
                   const float* __restrict__ gamma, const float* __restrict__ beta,
                   const float* __restrict__ counts, int cntIdx, int D, int doL2,
                   bf16* __restrict__ out) {
  __shared__ float red[256];
  const long long row = blockIdx.x;
  const int t = threadIdx.x;
  const float cnt = counts[cntIdx];
  const int nc = D >> 8;   // D/256 (1 or 2)
  float y[2];
  float local = 0.0f;
#pragma unroll 2
  for (int i = 0; i < nc; ++i) {
    int c = t + (i << 8);
    float x = pre[row * D + c];
    float mean = sums[c] / cnt;
    float var  = sums[D + c] / cnt - mean * mean;
    float v = (x - mean) * rsqrtf(var + 1e-5f) * gamma[c] + beta[c];
    v = fmaxf(v, 0.0f);
    y[i] = v;
    local += v * v;
  }
  if (doL2) {
    red[t] = local;
    __syncthreads();
    for (int s = 128; s > 0; s >>= 1) {
      if (t < s) red[t] += red[t + s];
      __syncthreads();
    }
    float scale = 1.0f / fmaxf(sqrtf(red[0]), 1e-12f);
#pragma unroll 2
    for (int i = 0; i < nc; ++i) y[i] *= scale;
  }
#pragma unroll 2
  for (int i = 0; i < nc; ++i)
    out[row * D + t + (i << 8)] = (bf16)y[i];
}

// ---------------------------------------------------------------------------
// Final BN apply + ReLU + query-row mask -> f32 output (D == 256)
// ---------------------------------------------------------------------------
__global__ __launch_bounds__(256)
void bn_apply_final(const float* __restrict__ pre, const float* __restrict__ sums,
                    const float* __restrict__ gamma, const float* __restrict__ beta,
                    const float* __restrict__ counts, const int* __restrict__ a_lens,
                    float* __restrict__ out) {
  const long long row = blockIdx.x;
  const int t = threadIdx.x;
  const int smp = (int)(row / LA);
  const int pos = (int)(row - (long long)smp * LA);
  const float m = (pos < a_lens[smp]) ? 1.0f : 0.0f;
  const float cnt = counts[0];
  float x = pre[row * DQ + t];
  float mean = sums[t] / cnt;
  float var  = sums[DQ + t] / cnt - mean * mean;
  float v = (x - mean) * rsqrtf(var + 1e-5f) * gamma[t] + beta[t];
  v = fmaxf(v, 0.0f);
  out[row * DQ + t] = v * m;
}

// ---------------------------------------------------------------------------
// Masked softmax over key axis, one wave per query row -> bf16 weights
// ---------------------------------------------------------------------------
__global__ __launch_bounds__(256)
void softmax_rows(const float* __restrict__ scores, const int* __restrict__ b_lens,
                  bf16* __restrict__ w) {
  const int wave = threadIdx.x >> 5;
  const int lane = threadIdx.x & 31;
  const long long row = (long long)blockIdx.x * 8 + wave;
  const int smp = (int)(row / LA);
  const int blen = b_lens[smp];
  const float* sr = scores + row * LB;

  float vals[LB / 32];
  float mx = -3.0e38f;
#pragma unroll
  for (int j = 0; j < LB / 32; ++j) {
    int col = lane + j * 32;
    float v = sr[col] * SCALER;
    if (col >= blen) v = -1.0e30f;
    vals[j] = v;
    mx = fmaxf(mx, v);
  }
#pragma unroll
  for (int m = 16; m >= 1; m >>= 1) mx = fmaxf(mx, __shfl_xor(mx, m, 32));

  float sum = 0.0f;
#pragma unroll
  for (int j = 0; j < LB / 32; ++j) {
    float e = __expf(vals[j] - mx);
    vals[j] = e;
    sum += e;
  }
#pragma unroll
  for (int m = 16; m >= 1; m >>= 1) sum += __shfl_xor(sum, m, 32);

  float inv = 1.0f / sum;
  bf16* wr = w + row * LB;
#pragma unroll
  for (int j = 0; j < LB / 32; ++j)
    wr[lane + j * 32] = (bf16)(vals[j] * inv);
}

// ---------------------------------------------------------------------------
// Orchestration
// ---------------------------------------------------------------------------
static inline dim3 grid1(long long n) { return dim3((unsigned)((n + 255) / 256)); }

extern "C" void kernel_launch(void* const* d_in, const int* in_sizes, int n_in,
                              void* d_out, int out_size, void* d_ws, size_t ws_size,
                              hipStream_t stream) {
  (void)in_sizes; (void)n_in; (void)out_size; (void)ws_size;

  const float* A      = (const float*)d_in[0];
  const float* B      = (const float*)d_in[1];
  const int*   a_lens = (const int*)d_in[2];
  const int*   b_lens = (const int*)d_in[3];
  const float* Wq = (const float*)d_in[4],  *bq = (const float*)d_in[5];
  const float* gq = (const float*)d_in[6],  *betaq = (const float*)d_in[7];
  const float* Wk = (const float*)d_in[8],  *bk = (const float*)d_in[9];
  const float* gk = (const float*)d_in[10], *betak = (const float*)d_in[11];
  const float* Wv = (const float*)d_in[12], *bv = (const float*)d_in[13];
  const float* gv = (const float*)d_in[14], *betav = (const float*)d_in[15];
  const float* Wf = (const float*)d_in[16], *bf_ = (const float*)d_in[17];
  const float* gf = (const float*)d_in[18], *betaf = (const float*)d_in[19];
  float* out = (float*)d_out;

  // ---- workspace carving ----
  char* ws = (char*)d_ws;
  size_t off = 0;
  auto alloc = [&](size_t bytes) -> char* {
    char* p = ws + off;
    off += (bytes + 255) & ~(size_t)255;
    return p;
  };
  bf16*  A16   = (bf16*)alloc((size_t)MA * DQ   * 2);
  bf16*  B16   = (bf16*)alloc((size_t)MB * DKVP * 2);
  bf16*  Wq16  = (bf16*)alloc((size_t)DQK * DQ   * 2);
  bf16*  Wk16  = (bf16*)alloc((size_t)DQK * DKVP * 2);
  bf16*  Wv16  = (bf16*)alloc((size_t)DV  * DKVP * 2);
  bf16*  Wf16  = (bf16*)alloc((size_t)DQ  * DQK  * 2);
  float* q_pre = (float*)alloc((size_t)MA * DQK * 4);
  float* k_pre = (float*)alloc((size_t)MB * DQK * 4);
  float* v_pre = (float*)alloc((size_t)MB * DV  * 4);
  bf16*  q16   = (bf16*)alloc((size_t)MA * DQK * 2);
  bf16*  k16   = (bf16*)alloc((size_t)MB * DQK * 2);
  bf16*  v16   = (bf16*)alloc((size_t)MB * DV  * 2);
  float* stats = (float*)alloc(4 * 1024 * 4);   // 4 regions of [2*512] f32
  float* counts= (float*)alloc(256);
  float* zbias = (float*)alloc(512 * 4);        // zero bias for score/wv GEMMs
  // aliases over dead pre-activation buffers
  float* scores = q_pre;                                   // S*La*Lb f32
  bf16*  w16    = (bf16*)k_pre;                            // S*La*Lb bf16
  float* wv     = v_pre;                                   // S*La*Dv f32
  bf16*  wv16   = (bf16*)((char*)v_pre + (size_t)MA * DV * 4);
  float* f_pre  = (float*)((char*)v_pre + (size_t)MA * DV * 4 + (size_t)MA * DV * 2);

  float* statsq = stats;
  float* statsk = stats + 1024;
  float* statsv = stats + 2048;
  float* statsf = stats + 3072;

  // ---- phase 0: conversions + counts + zeroing ----
  convert_pad_bf16<<<grid1((long long)MA * DQ),   256, 0, stream>>>(A,  A16,  MA, DQ,  DQ);
  convert_pad_bf16<<<grid1((long long)MB * DKVP), 256, 0, stream>>>(B,  B16,  MB, DKV, DKVP);
  convert_pad_bf16<<<grid1((long long)DQK * DQ),  256, 0, stream>>>(Wq, Wq16, DQK, DQ,  DQ);
  convert_pad_bf16<<<grid1((long long)DQK * DKVP),256, 0, stream>>>(Wk, Wk16, DQK, DKV, DKVP);
  convert_pad_bf16<<<grid1((long long)DV  * DKVP),256, 0, stream>>>(Wv, Wv16, DV,  DKV, DKVP);
  convert_pad_bf16<<<grid1((long long)DQ  * DQK), 256, 0, stream>>>(Wf, Wf16, DQ,  DQK, DQK);
  lens_counts<<<1, 256, 0, stream>>>(a_lens, b_lens, counts);
  zero_f32<<<18, 256, 0, stream>>>(stats, 4096 + 512);   // covers stats, counts and zbias[0..448)

  // ---- phase 1: projections (NT GEMMs) ----
  gemm_bf16_wmma<false><<<dim3(DQK / 128, MA / 128, 1), 256, 0, stream>>>(
      A16, Wq16, bq, q_pre, DQ, DQ, DQ, DQK, 0, 0, 0);
  gemm_bf16_wmma<false><<<dim3(DQK / 128, MB / 128, 1), 256, 0, stream>>>(
      B16, Wk16, bk, k_pre, DKVP, DKVP, DKVP, DQK, 0, 0, 0);
  gemm_bf16_wmma<false><<<dim3(DV / 128, MB / 128, 1), 256, 0, stream>>>(
      B16, Wv16, bv, v_pre, DKVP, DKVP, DKVP, DV, 0, 0, 0);

  // ---- phase 2: masked BN + ReLU + L2 -> bf16 q/k/v ----
  bn_stats<<<MA / 128, 256, 0, stream>>>(q_pre, a_lens, LA, DQK, statsq);
  bn_apply_bf16<<<MA, 256, 0, stream>>>(q_pre, statsq, gq, betaq, counts, 0, DQK, 1, q16);
  bn_stats<<<MB / 128, 256, 0, stream>>>(k_pre, b_lens, LB, DQK, statsk);
  bn_apply_bf16<<<MB, 256, 0, stream>>>(k_pre, statsk, gk, betak, counts, 1, DQK, 1, k16);
  bn_stats<<<MB / 128, 256, 0, stream>>>(v_pre, b_lens, LB, DV, statsv);
  bn_apply_bf16<<<MB, 256, 0, stream>>>(v_pre, statsv, gv, betav, counts, 1, DV, 1, v16);

  // ---- phase 3: scores = q k^T (batched NT), masked softmax -> bf16 w ----
  gemm_bf16_wmma<false><<<dim3(LB / 128, LA / 128, S_DIM), 256, 0, stream>>>(
      q16, k16, zbias, scores, DQK, DQK, DQK, LB,
      (long long)LA * DQK, (long long)LB * DQK, (long long)LA * LB);
  softmax_rows<<<MA / 8, 256, 0, stream>>>(scores, b_lens, w16);

  // ---- phase 4: wv = w v (batched NN) ----
  gemm_bf16_wmma<true><<<dim3(DV / 128, LA / 128, S_DIM), 256, 0, stream>>>(
      w16, v16, zbias, wv, LB, LB, DV, DV,
      (long long)LA * LB, (long long)LB * DV, (long long)LA * DV);
  convert_pad_bf16<<<grid1((long long)MA * DV), 256, 0, stream>>>(wv, wv16, MA, DV, DV);

  // ---- phase 5: final projection (NT) + BN + ReLU + row mask ----
  gemm_bf16_wmma<false><<<dim3(DQ / 128, MA / 128, 1), 256, 0, stream>>>(
      wv16, Wf16, bf_, f_pre, DQK, DQK, DQK, DQ, 0, 0, 0);
  bn_stats<<<MA / 128, 256, 0, stream>>>(f_pre, a_lens, LA, DQ, statsf);
  bn_apply_final<<<MA, 256, 0, stream>>>(f_pre, statsf, gf, betaf, counts, a_lens, out);
}
